// Model_29360396435655
// MI455X (gfx1250) — matrix operand order repaired and checked by reading, not compiled, not run on hardware
//
#include <hip/hip_runtime.h>
#include <hip/hip_bf16.h>
#include <math.h>

typedef __attribute__((ext_vector_type(16))) __bf16 v16bf;
typedef __attribute__((ext_vector_type(8)))  float  v8f;

static constexpr int Bn  = 32;
static constexpr int LL  = 512;

#define DEV static __device__ __forceinline__

DEV unsigned short f32_to_bf16(float f) {
  unsigned u = __builtin_bit_cast(unsigned, f);
  u += 0x7FFFu + ((u >> 16) & 1u);          // round-to-nearest-even
  return (unsigned short)(u >> 16);
}
DEV __bf16 us_to_bf(unsigned short h) { return __builtin_bit_cast(__bf16, h); }
DEV float gelu_exact(float x) { return 0.5f * x * (1.0f + erff(x * 0.70710678118654752f)); }

// ---------------------------------------------------------------------------
// Channel-first 1x1-conv GEMM:  Y[b, yOff+o, l] = act( sum_c W[o,c] * Xin + bias[o] )
// where Xin = X[b, xOff+c, l] (optionally * X2[b, c, l]).
// Tile: 64 (O) x 64 (L) per workgroup, K-step 32 via LDS, bf16 WMMA, f32 acc.
// Grid: (LL/64, O/64, nb). Block: 128 threads (4 waves, 2x2 wave grid).
// ---------------------------------------------------------------------------
template <bool GELU>
__global__ __launch_bounds__(128)
void gemm_cf_wmma(const float* __restrict__ X, long xBatch, int xChanOff,
                  const float* __restrict__ X2, long x2Batch,
                  const float* __restrict__ W, const float* __restrict__ Bias,
                  float* __restrict__ Y, long yBatch, int yChanOff,
                  int O, int C)
{
  __shared__ unsigned short lds_a[64][34];   // [o'][k']  (A tile, bf16 bits)
  __shared__ unsigned short lds_bt[64][34];  // [l'][k']  (B tile, transposed)

  const int tid  = threadIdx.x;
  const int lane = tid & 31;
  const int wave = tid >> 5;
  const int wm   = (wave >> 1) * 32;   // wave row offset within 64x64 tile
  const int wn   = (wave & 1) * 32;    // wave col offset
  const int b    = blockIdx.z;
  const int l0   = blockIdx.x * 64;
  const int o0   = blockIdx.y * 64;

  const float* Xb  = X + (long)b * xBatch + (long)xChanOff * LL;
  const float* X2b = X2 ? (X2 + (long)b * x2Batch) : nullptr;

  v8f acc[2][2] = {};

  const int arow = tid >> 3;          // 0..15
  const int acol = (tid & 7) * 4;     // 0..28
  const int brow = tid >> 4;          // 0..7
  const int bcol = (tid & 15) * 4;    // 0..60

  for (int c0 = 0; c0 < C; c0 += 32) {
    // ---- stage A tile: W[o0..o0+63, c0..c0+31] -> bf16 LDS
#pragma unroll
    for (int rr = 0; rr < 64; rr += 16) {
      const float4 wv = *(const float4*)(W + (long)(o0 + arow + rr) * C + (c0 + acol));
      lds_a[arow + rr][acol + 0] = f32_to_bf16(wv.x);
      lds_a[arow + rr][acol + 1] = f32_to_bf16(wv.y);
      lds_a[arow + rr][acol + 2] = f32_to_bf16(wv.z);
      lds_a[arow + rr][acol + 3] = f32_to_bf16(wv.w);
    }
    // ---- stage B tile: X[b, c0..c0+31, l0..l0+63] -> transposed bf16 LDS
#pragma unroll
    for (int jj = 0; jj < 32; jj += 8) {
      float4 xv = *(const float4*)(Xb + (long)(c0 + brow + jj) * LL + (l0 + bcol));
      if (X2b) {
        const float4 mv = *(const float4*)(X2b + (long)(c0 + brow + jj) * LL + (l0 + bcol));
        xv.x *= mv.x; xv.y *= mv.y; xv.z *= mv.z; xv.w *= mv.w;
      }
      lds_bt[bcol + 0][brow + jj] = f32_to_bf16(xv.x);
      lds_bt[bcol + 1][brow + jj] = f32_to_bf16(xv.y);
      lds_bt[bcol + 2][brow + jj] = f32_to_bf16(xv.z);
      lds_bt[bcol + 3][brow + jj] = f32_to_bf16(xv.w);
    }
    if (c0 + 32 < C)   // gfx1250 global_prefetch of next weight K-tile
      __builtin_prefetch(W + (long)(o0 + arow) * C + (c0 + 32 + acol), 0, 1);
    __syncthreads();

    // ---- WMMA over the K=32 tile (ISA 7.12.2 bf16 layouts)
    const int am  = lane & 15;
    const int akb = (lane >> 4) * 8;    // A: lanes 0-15 K{0-7,16-23}; 16-31 K{8-15,24-31}
    const int bkb = (lane >> 4) * 8;    // B: lanes 0-15 K{0-7,16-23}; 16-31 K{8-15,24-31}
#pragma unroll
    for (int mt = 0; mt < 2; ++mt) {
      v16bf afrag;
      const int r = wm + mt * 16 + am;
#pragma unroll
      for (int v = 0; v < 8; ++v) {
        const int k = (v < 4) ? (akb + 2 * v) : (16 + akb + 2 * (v - 4));
        afrag[2 * v + 0] = us_to_bf(lds_a[r][k + 0]);
        afrag[2 * v + 1] = us_to_bf(lds_a[r][k + 1]);
      }
#pragma unroll
      for (int nt = 0; nt < 2; ++nt) {
        v16bf bfrag;
        const int nc = wn + nt * 16 + am;
#pragma unroll
        for (int v = 0; v < 8; ++v) {
          const int k = (v < 4) ? (bkb + 2 * v) : (16 + bkb + 2 * (v - 4));
          bfrag[2 * v + 0] = us_to_bf(lds_bt[nc][k + 0]);
          bfrag[2 * v + 1] = us_to_bf(lds_bt[nc][k + 1]);
        }
        acc[mt][nt] = __builtin_amdgcn_wmma_f32_16x16x32_bf16(
            false, afrag, false, bfrag, (short)0, acc[mt][nt], false, false);
      }
    }
    __syncthreads();
  }

  // ---- epilogue: bias (+ GELU), scatter per C/D layout (VGPR j -> M=j or j+8)
  float* Yb = Y + (long)b * yBatch + (long)yChanOff * LL;
  const int dn  = lane & 15;
  const int dmo = (lane >> 4) * 8;
#pragma unroll
  for (int mt = 0; mt < 2; ++mt)
#pragma unroll
    for (int nt = 0; nt < 2; ++nt)
#pragma unroll
      for (int j = 0; j < 8; ++j) {
        const int o = o0 + wm + mt * 16 + dmo + j;
        const int l = l0 + wn + nt * 16 + dn;
        float val = acc[mt][nt][j] + Bias[o];
        if (GELU) val = gelu_exact(val);
        Yb[(long)o * LL + l] = val;
      }
}

// ---------------------------------------------------------------------------
// Channel-first LayerNorm: per (b,l), normalize over C channels.
// Grid: Bn*LL/256 blocks of 256.
// ---------------------------------------------------------------------------
__global__ void ln_cf_kernel(const float* __restrict__ X, long xBatch,
                             const float* __restrict__ Wg, const float* __restrict__ Bg,
                             float* __restrict__ Y, long yBatch, int C)
{
  const int idx = blockIdx.x * blockDim.x + threadIdx.x;  // over Bn*LL
  const int b = idx / LL, l = idx % LL;
  const float* xp = X + (long)b * xBatch + l;
  float s = 0.f, ss = 0.f;
  for (int c = 0; c < C; ++c) { const float v = xp[(long)c * LL]; s += v; ss += v * v; }
  const float mean = s / (float)C;
  const float var  = ss / (float)C - mean * mean;
  const float inv  = rsqrtf(var + 1e-6f);
  float* yp = Y + (long)b * yBatch + l;
  for (int c = 0; c < C; ++c) {
    const float v = xp[(long)c * LL];
    yp[(long)c * LL] = Wg[c] * ((v - mean) * inv) + Bg[c];
  }
}

// ---------------------------------------------------------------------------
// Depthwise conv k=3 pad=1 on a (Bn,64,LL) tensor, + bias + optional addend,
// written into Y at channel offset. Grid: Bn*64*LL/256.
// ---------------------------------------------------------------------------
__global__ void dw3_kernel(const float* __restrict__ X,
                           const float* __restrict__ W3, const float* __restrict__ Bg,
                           const float* __restrict__ Add, long addBatch, int addChanOff,
                           float* __restrict__ Y, long yBatch, int yChanOff)
{
  const long idx = (long)blockIdx.x * blockDim.x + threadIdx.x;
  const int l = (int)(idx % LL);
  const int c = (int)((idx / LL) % 64);
  const int b = (int)(idx / ((long)64 * LL));
  const float* xp = X + ((long)b * 64 + c) * LL;
  const float xm = (l > 0)      ? xp[l - 1] : 0.f;
  const float x0 = xp[l];
  const float xp1 = (l < LL - 1) ? xp[l + 1] : 0.f;
  float v = W3[c * 3 + 0] * xm + W3[c * 3 + 1] * x0 + W3[c * 3 + 2] * xp1 + Bg[c];
  if (Add) v += Add[(long)b * addBatch + (long)(addChanOff + c) * LL + l];
  Y[(long)b * yBatch + (long)(yChanOff + c) * LL + l] = v;
}

// ---------------------------------------------------------------------------
// DCN 1D bilinear sampling for ONE batch: t (C,LL), om (2*C*K, LL) -> out (C,LL).
// out[c,l] = sum_k bilinear(t[c], l + (k-(K-1)/2) + off[c,k,l]) * mask[c,k,l]
// ---------------------------------------------------------------------------
__global__ void dcn_sample_kernel(const float* __restrict__ T,
                                  const float* __restrict__ OM,
                                  float* __restrict__ Out, int C, int K)
{
  const int idx = blockIdx.x * blockDim.x + threadIdx.x;  // C*LL
  const int l = idx % LL, c = idx / LL;
  const float* tc = T + (long)c * LL;
  const int half = (K - 1) / 2;
  float acc = 0.f;
  for (int k = 0; k < K; ++k) {
    const float off  = OM[(long)(c * 2 * K + k) * LL + l];
    const float mask = OM[(long)(c * 2 * K + K + k) * LL + l];
    const float pos  = (float)l + (float)(k - half) + off;
    const float p0f  = floorf(pos);
    const float frac = pos - p0f;
    const int   p0   = (int)p0f;
    const float v0 = (p0 >= 0 && p0 < LL)           ? tc[p0]     : 0.f;
    const float v1 = (p0 + 1 >= 0 && p0 + 1 < LL)   ? tc[p0 + 1] : 0.f;
    acc += (v0 * (1.f - frac) + v1 * frac) * mask;
  }
  Out[idx] = acc;
}

// Out[b,c,l] = X[b,c,l] + S[c] * Yv[b,c,l], contiguous (Bn,C,LL)
__global__ void axpy_scale_kernel(const float* __restrict__ X, const float* __restrict__ S,
                                  const float* __restrict__ Yv, float* __restrict__ Out, int C)
{
  const long idx = (long)blockIdx.x * blockDim.x + threadIdx.x;
  const int c = (int)((idx / LL) % C);
  Out[idx] = X[idx] + S[c] * Yv[idx];
}

// ---------------------------------------------------------------------------
extern "C" void kernel_launch(void* const* d_in, const int* in_sizes, int n_in,
                              void* d_out, int out_size, void* d_ws, size_t ws_size,
                              hipStream_t stream)
{
  (void)in_sizes; (void)n_in; (void)out_size; (void)ws_size;
  auto IN = [&](int i) { return (const float*)d_in[i]; };

  // setup_inputs() dict order (recursive insertion order)
  const float* Xin        = IN(0);
  const float* layer_scale= IN(1);  const float* gamma2 = IN(2);
  const float* ln2_w = IN(3);       const float* ln2_b  = IN(4);
  const float* fc1_w = IN(5);       const float* fc1_b  = IN(6);
  const float* fc2_w = IN(7);       const float* fc2_b  = IN(8);
  const float* n1_w  = IN(9);       const float* n1_b   = IN(10);
  const float* a1_w  = IN(11);      const float* a1_b   = IN(12);
  const float* d1_om_w = IN(13);    const float* d1_om_b = IN(14);
  const float* d1_out_w = IN(15);   const float* d1_out_b = IN(16);
  const float* v1_w  = IN(17);      const float* v1_b   = IN(18);
  const float* v11_w = IN(19);      const float* v11_b  = IN(20);
  const float* v12_w = IN(21);      const float* v12_b  = IN(22);
  const float* c31_w = IN(23);      const float* c31_b  = IN(24);
  const float* n2_w  = IN(25);      const float* n2_b   = IN(26);
  const float* a2_w  = IN(27);      const float* a2_b   = IN(28);
  const float* d2_om_w = IN(29);    const float* d2_om_b = IN(30);
  const float* d2_out_w = IN(31);   const float* d2_out_b = IN(32);
  const float* v2_w  = IN(33);      const float* v2_b   = IN(34);
  const float* v21_w = IN(35);      const float* v21_b  = IN(36);
  const float* v22_w = IN(37);      const float* v22_b  = IN(38);
  const float* proj2_w = IN(39);    const float* proj2_b = IN(40);
  const float* c32_w = IN(41);      const float* c32_b  = IN(42);
  const float* n3_w  = IN(43);      const float* n3_b   = IN(44);
  const float* a3_w  = IN(45);      const float* a3_b   = IN(46);
  const float* d3_om_w = IN(47);    const float* d3_om_b = IN(48);
  const float* d3_out_w = IN(49);   const float* d3_out_b = IN(50);
  const float* v3_w  = IN(51);      const float* v3_b   = IN(52);
  const float* v31_w = IN(53);      const float* v31_b  = IN(54);
  const float* v32_w = IN(55);      const float* v32_b  = IN(56);
  const float* proj3_w = IN(57);    const float* proj3_b = IN(58);
  const float* c33_w = IN(59);      const float* c33_b  = IN(60);

  // workspace layout (floats)
  float* ws = (float*)d_ws;
  size_t ofs = 0;
  auto alloc = [&](size_t nfl) { float* p = ws + ofs; ofs += nfl; return p; };
  const size_t BL = (size_t)Bn * LL;
  float* xn  = alloc(256 * BL);
  float* tb  = alloc(192 * BL);   // gelu(dcn input), per-stage channel count
  float* ab  = alloc(192 * BL);   // dcn output "a"
  float* vb  = alloc(192 * BL);   // value branch
  float* dcn = alloc(192 * BL);   // sampled, pre out-proj
  float* om  = alloc((size_t)4224 * LL);  // per-batch offset/mask buffer
  float* t64 = alloc(64 * BL);
  float* prj = alloc(64 * BL);
  float* x1  = alloc(128 * BL);
  float* x1n = alloc(128 * BL);
  float* x2  = alloc(192 * BL);
  float* x2n = alloc(192 * BL);
  float* fg  = alloc(256 * BL);
  float* Rr  = alloc(256 * BL);
  float* yn  = alloc(256 * BL);
  float* hh  = alloc(1024 * BL);
  float* mm  = alloc(256 * BL);

  const dim3 blk(128, 1, 1);
  auto gemm = [&](bool gelu, const float* X, long xB, int xOff, const float* X2, long x2B,
                  const float* W, const float* bias, float* Y, long yB, int yOff,
                  int O, int C, int nb) {
    dim3 g(LL / 64, O / 64, nb);
    if (gelu) gemm_cf_wmma<true ><<<g, blk, 0, stream>>>(X, xB, xOff, X2, x2B, W, bias, Y, yB, yOff, O, C);
    else      gemm_cf_wmma<false><<<g, blk, 0, stream>>>(X, xB, xOff, X2, x2B, W, bias, Y, yB, yOff, O, C);
  };
  auto ln = [&](const float* X, long xB, const float* w, const float* b, float* Y, long yB, int C) {
    ln_cf_kernel<<<(Bn * LL) / 256, 256, 0, stream>>>(X, xB, w, b, Y, yB, C);
  };
  auto dw3 = [&](const float* X, const float* w3, const float* b, const float* add,
                 long addB, int addOff, float* Y, long yB, int yOff) {
    dw3_kernel<<<(Bn * 64 * LL) / 256, 256, 0, stream>>>(X, w3, b, add, addB, addOff, Y, yB, yOff);
  };
  auto dcn_stage = [&](int C, int K, const float* om_w, const float* om_b) {
    for (int b = 0; b < Bn; ++b) {
      gemm(false, tb + (size_t)b * C * LL, 0, 0, nullptr, 0, om_w, om_b,
           om, 0, 0, 2 * C * K, C, 1);
      dcn_sample_kernel<<<(C * LL) / 256, 256, 0, stream>>>(
          tb + (size_t)b * C * LL, om, dcn + (size_t)b * C * LL, C, K);
    }
  };

  // ======================= FGDM =======================
  ln(Xin, 256L * LL, n1_w, n1_b, xn, 256L * LL, 256);

  // ---- stage 1 (C=64, K=7)
  gemm(true, xn, 256L * LL, 0, nullptr, 0, a1_w, a1_b, tb, 64L * LL, 0, 64, 64, Bn);
  dcn_stage(64, 7, d1_om_w, d1_om_b);
  gemm(false, dcn, 64L * LL, 0, nullptr, 0, d1_out_w, d1_out_b, ab, 64L * LL, 0, 64, 64, Bn);   // a1
  gemm(false, xn, 256L * LL, 0, nullptr, 0, v1_w, v1_b, vb, 64L * LL, 0, 64, 64, Bn);           // v1
  gemm(false, ab, 64L * LL, 0, vb, 64L * LL, v11_w, v11_b, x1, 128L * LL, 64, 64, 64, Bn);      // mul1
  gemm(false, xn, 256L * LL, 64, nullptr, 0, v12_w, v12_b, t64, 64L * LL, 0, 64, 64, Bn);
  dw3(t64, c31_w, c31_b, ab, 64L * LL, 0, x1, 128L * LL, 0);                                    // x1a
  ln(x1, 128L * LL, n2_w, n2_b, x1n, 128L * LL, 128);

  // ---- stage 2 (C=128, K=9)
  gemm(true, x1n, 128L * LL, 0, nullptr, 0, a2_w, a2_b, tb, 128L * LL, 0, 128, 128, Bn);
  dcn_stage(128, 9, d2_om_w, d2_om_b);
  gemm(false, dcn, 128L * LL, 0, nullptr, 0, d2_out_w, d2_out_b, ab, 128L * LL, 0, 128, 128, Bn); // a2
  gemm(false, x1n, 128L * LL, 0, nullptr, 0, v2_w, v2_b, vb, 128L * LL, 0, 128, 128, Bn);         // v2
  gemm(false, ab, 128L * LL, 0, vb, 128L * LL, v21_w, v21_b, x2, 192L * LL, 64, 128, 128, Bn);    // mul2
  gemm(false, ab, 128L * LL, 0, nullptr, 0, proj2_w, proj2_b, prj, 64L * LL, 0, 64, 128, Bn);
  gemm(false, xn, 256L * LL, 128, nullptr, 0, v22_w, v22_b, t64, 64L * LL, 0, 64, 64, Bn);
  dw3(t64, c32_w, c32_b, prj, 64L * LL, 0, x2, 192L * LL, 0);                                     // x2a
  ln(x2, 192L * LL, n3_w, n3_b, x2n, 192L * LL, 192);

  // ---- stage 3 (C=192, K=11)
  gemm(true, x2n, 192L * LL, 0, nullptr, 0, a3_w, a3_b, tb, 192L * LL, 0, 192, 192, Bn);
  dcn_stage(192, 11, d3_om_w, d3_om_b);
  gemm(false, dcn, 192L * LL, 0, nullptr, 0, d3_out_w, d3_out_b, ab, 192L * LL, 0, 192, 192, Bn); // a3
  gemm(false, x2n, 192L * LL, 0, nullptr, 0, v3_w, v3_b, vb, 192L * LL, 0, 192, 192, Bn);         // v3
  gemm(false, ab, 192L * LL, 0, vb, 192L * LL, v31_w, v31_b, fg, 256L * LL, 64, 192, 192, Bn);    // mul3
  gemm(false, ab, 192L * LL, 0, nullptr, 0, proj3_w, proj3_b, prj, 64L * LL, 0, 64, 192, Bn);
  gemm(false, xn, 256L * LL, 192, nullptr, 0, v32_w, v32_b, t64, 64L * LL, 0, 64, 64, Bn);
  dw3(t64, c33_w, c33_b, prj, 64L * LL, 0, fg, 256L * LL, 0);                                     // x3a

  // ======================= block: residual + MLP =======================
  axpy_scale_kernel<<<(Bn * 256 * LL) / 256, 256, 0, stream>>>(Xin, layer_scale, fg, Rr, 256);
  ln(Rr, 256L * LL, ln2_w, ln2_b, yn, 256L * LL, 256);
  gemm(true, yn, 256L * LL, 0, nullptr, 0, fc1_w, fc1_b, hh, 1024L * LL, 0, 1024, 256, Bn);
  gemm(false, hh, 1024L * LL, 0, nullptr, 0, fc2_w, fc2_b, mm, 256L * LL, 0, 256, 1024, Bn);
  axpy_scale_kernel<<<(Bn * 256 * LL) / 256, 256, 0, stream>>>(Rr, gamma2, mm, (float*)d_out, 256);
}
